// MHA_45904610459991
// MI455X (gfx1250) — matrix operand order repaired and checked by reading, not compiled, not run on hardware
//
#include <hip/hip_runtime.h>

typedef unsigned short ushort_t;
typedef __attribute__((ext_vector_type(16))) __bf16        v16bf;
typedef __attribute__((ext_vector_type(8)))  float         v8f;
typedef __attribute__((ext_vector_type(4)))  unsigned int  u32x4;
typedef __attribute__((ext_vector_type(8)))  unsigned int  u32x8;

union V16BF { v16bf v; u32x4 q[2]; u32x8 o; };

static constexpr int D_MODEL = 1024;
static constexpr int N_HEADS = 16;
static constexpr int D_K     = 64;
static constexpr int BB      = 4;
static constexpr int SS      = 2048;
static constexpr int M_ROWS  = BB * SS;        // 8192
static constexpr int N_QKV   = 3 * D_MODEL;    // 3072

__device__ __forceinline__ ushort_t f2bf(float f) {
  unsigned u = __float_as_uint(f);
  unsigned r = u + 0x7fffu + ((u >> 16) & 1u);   // round-to-nearest-even
  return (ushort_t)(r >> 16);
}
__device__ __forceinline__ float bf2f(ushort_t u) {
  return __uint_as_float(((unsigned)u) << 16);
}

// ---------------------------------------------------------------------------
// fp32 -> bf16 conversion
// ---------------------------------------------------------------------------
__global__ void cvt_bf16_kernel(const float* __restrict__ src,
                                ushort_t* __restrict__ dst, int n) {
  int i = blockIdx.x * blockDim.x + threadIdx.x;
  if (i < n) dst[i] = f2bf(src[i]);
}

// ---------------------------------------------------------------------------
// TN GEMM: C[M,N] = A[M,K] * W[N,K]^T   (bf16 in, f32 accumulate)
// wave tile 32x64 (2x4 WMMA accumulators), block tile 128x128 (8 waves)
// ---------------------------------------------------------------------------
template <bool OUT_BF16>
__global__ __launch_bounds__(256, 1)
void gemm_tn_bf16(const ushort_t* __restrict__ A,
                  const ushort_t* __restrict__ W,
                  void* __restrict__ Cout,
                  int M, int N, int K) {
  const int lane = threadIdx.x & 31;
  const int wave = threadIdx.x >> 5;
  const int wm   = wave & 3;
  const int wn   = wave >> 2;
  const int m0   = blockIdx.y * 128 + wm * 32;
  const int n0   = blockIdx.x * 128 + wn * 64;
  const int half = lane >> 4;
  const int l16  = lane & 15;
  const int koff = half * 16;

  v8f acc[2][4];
#pragma unroll
  for (int im = 0; im < 2; ++im)
#pragma unroll
    for (int in = 0; in < 4; ++in)
      acc[im][in] = (v8f){0.f, 0.f, 0.f, 0.f, 0.f, 0.f, 0.f, 0.f};

  for (int kb = 0; kb < K; kb += 32) {
    V16BF af[2];
#pragma unroll
    for (int im = 0; im < 2; ++im) {
      const ushort_t* ap = A + (size_t)(m0 + im * 16 + l16) * K + kb + koff;
      af[im].q[0] = *(const u32x4*)ap;
      af[im].q[1] = *(const u32x4*)(ap + 8);
      __builtin_prefetch(ap + 32, 0, 3);           // next K-step, WGP-scope prefetch
    }
    V16BF bf[4];
#pragma unroll
    for (int in = 0; in < 4; ++in) {
      const ushort_t* bp = W + (size_t)(n0 + in * 16 + l16) * K + kb + half * 16;
      bf[in].o = *(const u32x8*)bp;
      __builtin_prefetch(bp + 32, 0, 3);
    }
#pragma unroll
    for (int im = 0; im < 2; ++im)
#pragma unroll
      for (int in = 0; in < 4; ++in)
        acc[im][in] = __builtin_amdgcn_wmma_f32_16x16x32_bf16(
            false, af[im].v, false, bf[in].v, (short)0, acc[im][in], false, false);
  }

#pragma unroll
  for (int im = 0; im < 2; ++im)
#pragma unroll
    for (int in = 0; in < 4; ++in)
#pragma unroll
      for (int r = 0; r < 8; ++r) {
        size_t row = (size_t)(m0 + im * 16 + half * 8 + r);
        size_t col = (size_t)(n0 + in * 16 + l16);
        float val  = acc[im][in][r];
        if (OUT_BF16) ((ushort_t*)Cout)[row * N + col] = f2bf(val);
        else          ((float*)Cout)[row * N + col]    = val;
      }
}

// ---------------------------------------------------------------------------
// RoPE on Q,K + head split; V transposed to [B,H,dk,S]
// one thread per (b, h, s, d2) pair slot
// ---------------------------------------------------------------------------
__global__ void rope_split_kernel(const ushort_t* __restrict__ qkv,
                                  const int* __restrict__ pos,
                                  ushort_t* __restrict__ Qh,
                                  ushort_t* __restrict__ Kh,
                                  ushort_t* __restrict__ Vt) {
  int t  = blockIdx.x * blockDim.x + threadIdx.x;   // 2^22 total
  int d2 = t & 31;
  int s  = (t >> 5) & (SS - 1);
  int h  = (t >> 16) & (N_HEADS - 1);
  int b  = t >> 20;

  size_t row = (size_t)(b * SS + s) * N_QKV;
  int    c   = h * D_K + 2 * d2;

  float q1 = bf2f(qkv[row + c]),            q2 = bf2f(qkv[row + c + 1]);
  float k1 = bf2f(qkv[row + D_MODEL + c]),  k2 = bf2f(qkv[row + D_MODEL + c + 1]);
  float v1 = bf2f(qkv[row + 2*D_MODEL + c]),v2 = bf2f(qkv[row + 2*D_MODEL + c + 1]);

  float p    = (float)pos[s];
  float freq = __powf(10000.0f, -(float)(2 * d2) * (1.0f / (float)D_K));
  float sn, cs;
  __sincosf(p * freq, &sn, &cs);

  size_t qidx = ((size_t)(b * N_HEADS + h) * SS + s) * D_K + 2 * d2;
  Qh[qidx]     = f2bf(q1 * cs - q2 * sn);
  Qh[qidx + 1] = f2bf(q1 * sn + q2 * cs);
  Kh[qidx]     = f2bf(k1 * cs - k2 * sn);
  Kh[qidx + 1] = f2bf(k1 * sn + k2 * cs);

  size_t vidx = ((size_t)(b * N_HEADS + h) * D_K + 2 * d2) * SS + s;
  Vt[vidx]      = f2bf(v1);
  Vt[vidx + SS] = f2bf(v2);
}

// ---------------------------------------------------------------------------
// Flash attention: each wave owns 16 query rows, iterates causal KV tiles of 32
// grid: x = S/128, y = B*H ; block = 256 threads (8 waves)
// ---------------------------------------------------------------------------
__global__ __launch_bounds__(256, 1)
void flash_attn_kernel(const ushort_t* __restrict__ Qh,
                       const ushort_t* __restrict__ Kh,
                       const ushort_t* __restrict__ Vt,
                       ushort_t* __restrict__ AO) {
  __shared__ ushort_t plds[8 * 512];               // 16x32 bf16 per wave

  const int lane = threadIdx.x & 31;
  const int wave = threadIdx.x >> 5;
  const int bh   = blockIdx.y;
  const int b    = bh >> 4;
  const int h    = bh & (N_HEADS - 1);
  const int qbase= blockIdx.x * 128 + wave * 16;
  const int half = lane >> 4;
  const int l16  = lane & 15;
  const int koff = half * 16;

  const ushort_t* Q  = Qh + (size_t)bh * SS * D_K;
  const ushort_t* Kp = Kh + (size_t)bh * SS * D_K;
  const ushort_t* Vp = Vt + (size_t)bh * D_K * SS;
  ushort_t* lds = plds + wave * 512;

  // preload Q fragments (dk = 64 -> two K-steps of 32)
  V16BF qf[2];
#pragma unroll
  for (int ks = 0; ks < 2; ++ks) {
    const ushort_t* qp = Q + (size_t)(qbase + l16) * D_K + ks * 32 + koff;
    qf[ks].q[0] = *(const u32x4*)qp;
    qf[ks].q[1] = *(const u32x4*)(qp + 8);
  }

  v8f O[4];
#pragma unroll
  for (int t = 0; t < 4; ++t) O[t] = (v8f){0.f,0.f,0.f,0.f,0.f,0.f,0.f,0.f};
  float mrow[8], lrow[8];
#pragma unroll
  for (int r = 0; r < 8; ++r) { mrow[r] = -__builtin_inff(); lrow[r] = 0.f; }

  const float scale = 0.125f;                      // 1/sqrt(64)

  for (int kv = 0; kv <= qbase + 15; kv += 32) {
    // prefetch next KV tile into WGP-near caches while this one computes
    __builtin_prefetch(Kp + (size_t)(kv + 32 + l16) * D_K, 0, 3);
    __builtin_prefetch(Vp + (size_t)(half * 32 + l16) * SS + kv + 32, 0, 3);

    // ---- scores: S(16x32) = Q(16x64) * K^T ----
    v8f sc[2];
    sc[0] = (v8f){0.f,0.f,0.f,0.f,0.f,0.f,0.f,0.f};
    sc[1] = (v8f){0.f,0.f,0.f,0.f,0.f,0.f,0.f,0.f};
#pragma unroll
    for (int ks = 0; ks < 2; ++ks) {
#pragma unroll
      for (int j = 0; j < 2; ++j) {
        V16BF kb;
        const ushort_t* kp = Kp + (size_t)(kv + j * 16 + l16) * D_K + ks * 32 + half * 16;
        kb.o = *(const u32x8*)kp;
        sc[j] = __builtin_amdgcn_wmma_f32_16x16x32_bf16(
            false, qf[ks].v, false, kb.v, (short)0, sc[j], false, false);
      }
    }
    // ---- causal mask + online softmax (row stats per VGPR slot r) ----
#pragma unroll
    for (int r = 0; r < 8; ++r) {
      int qrow = qbase + half * 8 + r;
      float s0 = sc[0][r] * scale; if (kv + l16      > qrow) s0 = -__builtin_inff();
      float s1 = sc[1][r] * scale; if (kv + 16 + l16 > qrow) s1 = -__builtin_inff();
      float mx = fmaxf(s0, s1);
      mx = fmaxf(mx, __shfl_xor(mx, 1));
      mx = fmaxf(mx, __shfl_xor(mx, 2));
      mx = fmaxf(mx, __shfl_xor(mx, 4));
      mx = fmaxf(mx, __shfl_xor(mx, 8));          // stays inside 16-lane half
      float nm    = fmaxf(mrow[r], mx);
      float alpha = __expf(mrow[r] - nm);
      float p0 = __expf(s0 - nm);
      float p1 = __expf(s1 - nm);
      sc[0][r] = p0; sc[1][r] = p1;
      float rs = p0 + p1;
      rs += __shfl_xor(rs, 1);
      rs += __shfl_xor(rs, 2);
      rs += __shfl_xor(rs, 4);
      rs += __shfl_xor(rs, 8);
      lrow[r] = lrow[r] * alpha + rs;
      mrow[r] = nm;
#pragma unroll
      for (int t = 0; t < 4; ++t) O[t][r] *= alpha;
    }
    // ---- C-layout -> A-layout transpose through wave-private LDS ----
#pragma unroll
    for (int j = 0; j < 2; ++j)
#pragma unroll
      for (int r = 0; r < 8; ++r)
        lds[(half * 8 + r) * 32 + j * 16 + l16] = f2bf(sc[j][r]);
    asm volatile("s_wait_dscnt 0" ::: "memory");   // LDS in-order per wave
    V16BF pf;
    const ushort_t* lp = lds + l16 * 32 + koff;
    pf.q[0] = *(const u32x4*)lp;
    pf.q[1] = *(const u32x4*)(lp + 8);
    // ---- O(16x64) += P(16x32) * V(32x64)  (V pre-transposed) ----
#pragma unroll
    for (int t = 0; t < 4; ++t) {
      V16BF vb;
      const ushort_t* vp = Vp + (size_t)(t * 16 + l16) * SS + kv + half * 16;
      vb.o = *(const u32x8*)vp;
      O[t] = __builtin_amdgcn_wmma_f32_16x16x32_bf16(
          false, pf.v, false, vb.v, (short)0, O[t], false, false);
    }
  }

  // ---- epilogue: normalize, write [B*S, H*dk] bf16 ----
#pragma unroll
  for (int r = 0; r < 8; ++r) {
    float inv  = 1.0f / lrow[r];
    size_t row = (size_t)(b * SS + qbase + half * 8 + r);
#pragma unroll
    for (int t = 0; t < 4; ++t) {
      size_t col = (size_t)(h * D_K + t * 16 + l16);
      AO[row * D_MODEL + col] = f2bf(O[t][r] * inv);
    }
  }
}

// ---------------------------------------------------------------------------
extern "C" void kernel_launch(void* const* d_in, const int* in_sizes, int n_in,
                              void* d_out, int out_size, void* d_ws, size_t ws_size,
                              hipStream_t stream) {
  const float* x   = (const float*)d_in[0];
  const float* Wq  = (const float*)d_in[1];
  const float* Wk  = (const float*)d_in[2];
  const float* Wv  = (const float*)d_in[3];
  const float* Wo  = (const float*)d_in[4];
  const int*   pos = (const int*)d_in[5];

  char* ws = (char*)d_ws;
  size_t off = 0;
  ushort_t* xb   = (ushort_t*)(ws + off); off += (size_t)M_ROWS * D_MODEL * 2;   // 16 MB
  ushort_t* wqkv = (ushort_t*)(ws + off); off += (size_t)N_QKV * D_MODEL * 2;    //  6 MB
  ushort_t* wo   = (ushort_t*)(ws + off); off += (size_t)D_MODEL * D_MODEL * 2;  //  2 MB
  ushort_t* qkvb = (ushort_t*)(ws + off); off += (size_t)M_ROWS * N_QKV * 2;     // 48 MB
  ushort_t* qh   = (ushort_t*)(ws + off); off += (size_t)M_ROWS * D_MODEL * 2;   // 16 MB
  ushort_t* kh   = (ushort_t*)(ws + off); off += (size_t)M_ROWS * D_MODEL * 2;   // 16 MB
  ushort_t* vt   = (ushort_t*)(ws + off); off += (size_t)M_ROWS * D_MODEL * 2;   // 16 MB
  ushort_t* ao   = (ushort_t*)(ws + off); off += (size_t)M_ROWS * D_MODEL * 2;   // 16 MB

  const int CT = 256;
  int nx = M_ROWS * D_MODEL;       // 8388608
  int nw = D_MODEL * D_MODEL;      // 1048576
  cvt_bf16_kernel<<<(nx + CT - 1) / CT, CT, 0, stream>>>(x,  xb,            nx);
  cvt_bf16_kernel<<<(nw + CT - 1) / CT, CT, 0, stream>>>(Wq, wqkv,          nw);
  cvt_bf16_kernel<<<(nw + CT - 1) / CT, CT, 0, stream>>>(Wk, wqkv + nw,     nw);
  cvt_bf16_kernel<<<(nw + CT - 1) / CT, CT, 0, stream>>>(Wv, wqkv + 2 * nw, nw);
  cvt_bf16_kernel<<<(nw + CT - 1) / CT, CT, 0, stream>>>(Wo, wo,            nw);

  // QKV projection: [8192,3072] = xb[8192,1024] @ wqkv[3072,1024]^T
  gemm_tn_bf16<true><<<dim3(N_QKV / 128, M_ROWS / 128), 256, 0, stream>>>(
      xb, wqkv, qkvb, M_ROWS, N_QKV, D_MODEL);

  // RoPE + head split + V transpose
  rope_split_kernel<<<(BB * N_HEADS * SS * 32) / CT, CT, 0, stream>>>(
      qkvb, pos, qh, kh, vt);

  // causal flash attention
  flash_attn_kernel<<<dim3(SS / 128, BB * N_HEADS), 256, 0, stream>>>(qh, kh, vt, ao);

  // output projection (f32 out): [8192,1024] = ao @ wo^T
  gemm_tn_bf16<false><<<dim3(D_MODEL / 128, M_ROWS / 128), 256, 0, stream>>>(
      ao, wo, (float*)d_out, M_ROWS, D_MODEL, D_MODEL);

  (void)in_sizes; (void)n_in; (void)out_size; (void)ws_size;
}
